// BiTrap_25048249270584
// MI455X (gfx1250) — matrix operand order repaired and checked by reading, not compiled, not run on hardware
//
#include <hip/hip_runtime.h>

// ---------------------------------------------------------------------------
// BiTrap fused kernel for MI455X (gfx1250, wave32, WMMA).
//
// Design (see analysis): per-row-independent recurrences => one fused kernel,
// each block owns a 32-row tile of N=16384 and runs the whole network with
// activations in LDS (f16) and all GEMMs as v_wmma_f32_16x16x32_f16.
// Weights are converted once to f16 into d_ws (L2-resident, ~2.7MB).
// ---------------------------------------------------------------------------

typedef _Float16 f16;
typedef __attribute__((ext_vector_type(16))) _Float16 v16h;
typedef __attribute__((ext_vector_type(8)))  _Float16 v8h;
typedef __attribute__((ext_vector_type(8)))  float    v8f;

#define NTOT   16384
#define ROWS   32        // rows (agents) per workgroup
#define NWAVES 8         // 256 threads = 8 wave32
#define GRUH   256

union V16u { v16h v; v8h h[2]; };

__device__ __forceinline__ float sigf(float x) { return 1.0f / (1.0f + __expf(-x)); }

__device__ __forceinline__ v8f wmma_f16(v16h a, v16h b, v8f c) {
  // D = A(16x32 f16) * B(32x16 f16) + C(16x16 f32)
  return __builtin_amdgcn_wmma_f32_16x16x32_f16(false, a, false, b, (short)0, c,
                                                false, false);
}

// A fragment (16x32 f16) from LDS, row-major with leading dim lda.
// Layout: lanes 0-15 hold K 0..15, lanes 16-31 hold K 16..31 (matches B).
__device__ __forceinline__ v16h load_a_frag(const f16* A, int lda, int mBase,
                                            int kBase, int lane) {
  const int half = lane >> 4;
  const int m    = mBase + (lane & 15);
  const f16* p   = A + (size_t)m * lda + kBase + half * 16;
  V16u u;
  u.h[0] = *(const v8h*)(p);        // K chunk 0 (16B)
  u.h[1] = *(const v8h*)(p + 8);    // K chunk 1 (16B)
  return u.v;
}

// B fragment (32x16 f16) from global weight W laid out [KOUT, K] row-major
// (PyTorch (o,i)); out = A @ W^T so B(k,n) = W[nBase+n, k].
// Lanes 0-15: K 0..15 contiguous (32B); lanes 16-31: K 16..31.
__device__ __forceinline__ v16h load_b_frag(const f16* W, int K, int nBase,
                                            int kBase, int lane) {
  const int half = lane >> 4;
  const int n    = nBase + (lane & 15);
  return *(const v16h*)(W + (size_t)n * K + kBase + half * 16);
}

// Generic dense: O[ROWS,KOUT] = act(A[ROWS,K] @ W^T + bias), O in LDS (f16).
__device__ __forceinline__ void dense(const f16* A, int lda, int K,
                                      const f16* W, const float* bias, int KOUT,
                                      f16* O, int ldo, bool relu,
                                      int wave, int lane) {
  const int mt = ROWS / 16, tiles = mt * (KOUT / 16);
  for (int t = wave; t < tiles; t += NWAVES) {
    const int mBase = (t % mt) * 16, nBase = (t / mt) * 16;
    v8f acc = {};
    for (int kb = 0; kb < K; kb += 32)
      acc = wmma_f16(load_a_frag(A, lda, mBase, kb, lane),
                     load_b_frag(W, K, nBase, kb, lane), acc);
    const int col = nBase + (lane & 15), half = lane >> 4;
    const float bv = bias ? bias[col] : 0.0f;
#pragma unroll
    for (int r = 0; r < 8; ++r) {
      float v = acc[r] + bv;
      if (relu) v = fmaxf(v, 0.0f);
      O[(size_t)(mBase + half * 8 + r) * ldo + col] = (f16)v;
    }
  }
}

// Dense writing f32 straight to global output (row base added).
__device__ __forceinline__ void dense_outf32(const f16* A, int lda, int K,
                                             const f16* W, const float* bias,
                                             int KOUT, float* O, int ldo,
                                             int rowBase, int wave, int lane) {
  const int mt = ROWS / 16, tiles = mt * (KOUT / 16);
  for (int t = wave; t < tiles; t += NWAVES) {
    const int mBase = (t % mt) * 16, nBase = (t / mt) * 16;
    v8f acc = {};
    for (int kb = 0; kb < K; kb += 32)
      acc = wmma_f16(load_a_frag(A, lda, mBase, kb, lane),
                     load_b_frag(W, K, nBase, kb, lane), acc);
    const int col = nBase + (lane & 15), half = lane >> 4;
    const float bv = bias[col];
#pragma unroll
    for (int r = 0; r < 8; ++r)
      O[(size_t)(rowBase + mBase + half * 8 + r) * ldo + col] = acc[r] + bv;
  }
}

// Full GRU cell step (PyTorch math), gates fused in accumulator registers.
__device__ __forceinline__ void gru_step(const f16* X, int KX, const f16* H,
                                         const f16* Wih, const f16* Whh,
                                         const float* bih, const float* bhh,
                                         f16* Hout, int wave, int lane) {
  const int mt = ROWS / 16;
  for (int t = wave; t < mt * (GRUH / 16); t += NWAVES) {
    const int mBase = (t % mt) * 16, cBase = (t / mt) * 16;
    v8f ar = {}, az = {}, an = {}, hr = {}, hz = {}, hn = {};
    for (int kb = 0; kb < KX; kb += 32) {
      v16h a = load_a_frag(X, KX, mBase, kb, lane);
      ar = wmma_f16(a, load_b_frag(Wih, KX, cBase,       kb, lane), ar);
      az = wmma_f16(a, load_b_frag(Wih, KX, 256 + cBase, kb, lane), az);
      an = wmma_f16(a, load_b_frag(Wih, KX, 512 + cBase, kb, lane), an);
    }
    for (int kb = 0; kb < GRUH; kb += 32) {
      v16h a = load_a_frag(H, GRUH, mBase, kb, lane);
      hr = wmma_f16(a, load_b_frag(Whh, GRUH, cBase,       kb, lane), hr);
      hz = wmma_f16(a, load_b_frag(Whh, GRUH, 256 + cBase, kb, lane), hz);
      hn = wmma_f16(a, load_b_frag(Whh, GRUH, 512 + cBase, kb, lane), hn);
    }
    const int col = cBase + (lane & 15), half = lane >> 4;
    const float bir = bih[col], biz = bih[256 + col], bin = bih[512 + col];
    const float bhr = bhh[col], bhz = bhh[256 + col], bhn = bhh[512 + col];
#pragma unroll
    for (int r = 0; r < 8; ++r) {
      const int m = mBase + half * 8 + r;
      float rg = sigf(ar[r] + bir + hr[r] + bhr);
      float zg = sigf(az[r] + biz + hz[r] + bhz);
      float ng = tanhf(an[r] + bin + rg * (hn[r] + bhn));
      float hv = (float)H[(size_t)m * GRUH + col];
      Hout[(size_t)m * GRUH + col] = (f16)((1.0f - zg) * ng + zg * hv);
    }
  }
}

// Backward GRU step with PRECOMPUTED gh = h_last @ Whh^T + bhh (reference's
// "bug": hidden input is constant h_last) -> only the K=64 input GEMM remains.
__device__ __forceinline__ void gru_step_pregh(const f16* X, const f16* GHp,
                                               const f16* Hc, const f16* Wih,
                                               const float* bih, f16* Out,
                                               int wave, int lane) {
  const int mt = ROWS / 16;
  for (int t = wave; t < mt * (GRUH / 16); t += NWAVES) {
    const int mBase = (t % mt) * 16, cBase = (t / mt) * 16;
    v8f ar = {}, az = {}, an = {};
    for (int kb = 0; kb < 64; kb += 32) {
      v16h a = load_a_frag(X, 64, mBase, kb, lane);
      ar = wmma_f16(a, load_b_frag(Wih, 64, cBase,       kb, lane), ar);
      az = wmma_f16(a, load_b_frag(Wih, 64, 256 + cBase, kb, lane), az);
      an = wmma_f16(a, load_b_frag(Wih, 64, 512 + cBase, kb, lane), an);
    }
    const int col = cBase + (lane & 15), half = lane >> 4;
    const float bir = bih[col], biz = bih[256 + col], bin = bih[512 + col];
#pragma unroll
    for (int r = 0; r < 8; ++r) {
      const int m = mBase + half * 8 + r;
      float ghr = (float)GHp[(size_t)m * 768 + col];
      float ghz = (float)GHp[(size_t)m * 768 + 256 + col];
      float ghn = (float)GHp[(size_t)m * 768 + 512 + col];
      float rg = sigf(ar[r] + bir + ghr);
      float zg = sigf(az[r] + biz + ghz);
      float ng = tanhf(an[r] + bin + rg * ghn);
      float hv = (float)Hc[(size_t)m * GRUH + col];
      Out[(size_t)m * GRUH + col] = (f16)((1.0f - zg) * ng + zg * hv);
    }
  }
}

struct Args {
  const float *x, *y, *e, *h0x, *h0y;
  const float *fcxw, *fcxb, *fcyw, *fcyb;
  const f16 *gxwih, *gxwhh; const float *gxbih, *gxbhh;
  const f16 *gywih, *gywhh; const float *gybih, *gybhh;
  const f16 *pr0w; const float *pr0b; const f16 *pr1w; const float *pr1b;
  const f16 *pr2w; const float *pr2b;
  const f16 *rc0w; const float *rc0b; const f16 *rc1w; const float *rc1b;
  const f16 *rc2w; const float *rc2b;
  const f16 *gl0w; const float *gl0b; const f16 *gl1w; const float *gl1b;
  const float *gl2w, *gl2b;
  const f16 *fcfw; const float *fcfb;
  const f16 *fc2w; const float *fc2b;
  const f16 *fc3w; const float *fc3b;
  const float *fc5w, *fc5b;
  const f16 *fwih, *fwhh; const float *fbih, *fbhh;
  const f16 *bwih, *bwhh; const float *bbih, *bbhh;
  const float *fc6w, *fc6b;
  float *out_p, *out_q, *out_g, *out_b;
};

// Dynamic LDS layout (f16 elements unless noted):
//  D0   [32*256]   D1 [32*256]    : recurrence double buffer / bh
//  CAT  [32*512]                  : concat(h, hy)
//  XZ   [32*288]                  : concat(h, z)
//  S1,S2[32*256]                  : MLP ping-pong / emb / f buffer / b buffer
//  GH   [32*768]                  : precomputed backward gh
//  PART [12*32*2] f32             : fc6 forward-half partials
//  G2   [32*2]    f32             : g / out scratch
#define OFF_D0  0
#define OFF_D1  8192
#define OFF_CAT 16384
#define OFF_XZ  32768
#define OFF_S1  41984
#define OFF_S2  50176
#define OFF_GH  58368
#define SM_F16_HALVES 82944
#define SM_BYTES (SM_F16_HALVES * 2 + (12 * 32 * 2 + 32 * 2) * 4)

__global__ __launch_bounds__(256) void bitrap_fused(Args A) {
  extern __shared__ char smraw[];
  f16*   SM   = (f16*)smraw;
  f16*   D0   = SM + OFF_D0;
  f16*   D1   = SM + OFF_D1;
  f16*   CAT  = SM + OFF_CAT;
  f16*   XZ   = SM + OFF_XZ;
  f16*   S1   = SM + OFF_S1;
  f16*   S2   = SM + OFF_S2;
  f16*   GH   = SM + OFF_GH;
  float* PART = (float*)(smraw + SM_F16_HALVES * 2);
  float* G2   = PART + 12 * 32 * 2;

  const int tid = threadIdx.x, lane = tid & 31, wave = tid >> 5;
  const int rowBase = blockIdx.x * ROWS;

  // ---- encode x ---------------------------------------------------------
  for (int i = tid; i < ROWS * GRUH; i += 256) {
    int r = i >> 8, c = i & 255;
    D0[i] = (f16)A.h0x[(size_t)(rowBase + r) * GRUH + c];
  }
  __syncthreads();
  f16 *hc = D0, *hn = D1;
  for (int t = 0; t < 8; ++t) {
    for (int i = tid; i < ROWS * 64; i += 256) {   // emb = lin(fcx, x_t)
      int r = i >> 6, o = i & 63;
      float x0 = A.x[(size_t)(0 * 8 + t) * NTOT + rowBase + r];
      float x1 = A.x[(size_t)(1 * 8 + t) * NTOT + rowBase + r];
      S1[i] = (f16)(x0 * A.fcxw[o * 2] + x1 * A.fcxw[o * 2 + 1] + A.fcxb[o]);
    }
    __syncthreads();
    gru_step(S1, 64, hc, A.gxwih, A.gxwhh, A.gxbih, A.gxbhh, hn, wave, lane);
    __syncthreads();
    f16* tm = hc; hc = hn; hn = tm;
  }
  for (int i = tid; i < ROWS * GRUH; i += 256) {   // CAT[:, :256] = h
    int r = i >> 8, c = i & 255;
    CAT[(size_t)r * 512 + c] = hc[i];
  }
  __syncthreads();

  // ---- prior MLP: 256 ->128 relu ->64 relu ->64 -> p --------------------
  dense(CAT, 512, 256, A.pr0w, A.pr0b, 128, S1, 128, true, wave, lane);
  __syncthreads();
  dense(S1, 128, 128, A.pr1w, A.pr1b, 64, S2, 64, true, wave, lane);
  __syncthreads();
  dense_outf32(S2, 64, 64, A.pr2w, A.pr2b, 64, A.out_p, 64, rowBase, wave, lane);
  __syncthreads();

  // ---- encode y ---------------------------------------------------------
  for (int i = tid; i < ROWS * GRUH; i += 256) {
    int r = i >> 8, c = i & 255;
    D0[i] = (f16)A.h0y[(size_t)(rowBase + r) * GRUH + c];
  }
  __syncthreads();
  hc = D0; hn = D1;
  for (int t = 0; t < 12; ++t) {
    for (int i = tid; i < ROWS * 64; i += 256) {
      int r = i >> 6, o = i & 63;
      float y0 = A.y[(size_t)(0 * 12 + t) * NTOT + rowBase + r];
      float y1 = A.y[(size_t)(1 * 12 + t) * NTOT + rowBase + r];
      S1[i] = (f16)(y0 * A.fcyw[o * 2] + y1 * A.fcyw[o * 2 + 1] + A.fcyb[o]);
    }
    __syncthreads();
    gru_step(S1, 64, hc, A.gywih, A.gywhh, A.gybih, A.gybhh, hn, wave, lane);
    __syncthreads();
    f16* tm = hc; hc = hn; hn = tm;
  }
  for (int i = tid; i < ROWS * GRUH; i += 256) {   // CAT[:, 256:] = hy
    int r = i >> 8, c = i & 255;
    CAT[(size_t)r * 512 + 256 + c] = hc[i];
  }
  __syncthreads();

  // ---- recognition MLP: 512 ->256 ->128 ->64 -> q, z, XZ ---------------
  dense(CAT, 512, 512, A.rc0w, A.rc0b, 256, S1, 256, true, wave, lane);
  __syncthreads();
  dense(S1, 256, 256, A.rc1w, A.rc1b, 128, S2, 128, true, wave, lane);
  __syncthreads();
  dense(S2, 128, 128, A.rc2w, A.rc2b, 64, D0, 64, false, wave, lane);  // q (f16)
  __syncthreads();
  for (int i = tid; i < ROWS * 64; i += 256) {
    int r = i >> 6, c = i & 63;
    A.out_q[(size_t)(rowBase + r) * 64 + c] = (float)D0[i];
  }
  for (int i = tid; i < ROWS * GRUH; i += 256) {   // XZ[:, :256] = h
    int r = i >> 8, c = i & 255;
    XZ[(size_t)r * 288 + c] = CAT[(size_t)r * 512 + c];
  }
  for (int i = tid; i < ROWS * 32; i += 256) {     // z = mu + sigma * e
    int r = i >> 5, j = i & 31;
    float z = (float)D0[(size_t)r * 64 + j] +
              (float)D0[(size_t)r * 64 + 32 + j] * A.e[(size_t)(rowBase + r) * 32 + j];
    XZ[(size_t)r * 288 + 256 + j] = (f16)z;
  }
  __syncthreads();

  // ---- goal MLP: 288 ->128 ->64 ->2 ; fwd_h0 = fc2(xz); f0 = fcf(xz) ----
  dense(XZ, 288, 288, A.gl0w, A.gl0b, 128, S1, 128, true, wave, lane);
  __syncthreads();
  dense(S1, 128, 128, A.gl1w, A.gl1b, 64, S2, 64, true, wave, lane);
  __syncthreads();
  if (tid < ROWS * 2) {                            // goal final 64->2
    int r = tid >> 1, c = tid & 1;
    float s = A.gl2b[c];
    for (int k = 0; k < 64; ++k) s += (float)S2[(size_t)r * 64 + k] * A.gl2w[c * 64 + k];
    A.out_g[(size_t)(rowBase + r) * 2 + c] = s;
    G2[tid] = s;
  }
  __syncthreads();
  dense(XZ, 288, 288, A.fc2w, A.fc2b, 256, D1, 256, false, wave, lane);  // fwd_h0
  __syncthreads();
  dense(XZ, 288, 288, A.fcfw, A.fcfb, 64, S1, 64, false, wave, lane);    // f0
  __syncthreads();

  // ---- forward rollout (12 steps) --------------------------------------
  hc = D1; hn = D0;
  for (int t = 0; t < 12; ++t) {
    gru_step(S1, 64, hc, A.fwih, A.fwhh, A.fbih, A.fbhh, hn, wave, lane);
    __syncthreads();
    if (tid < ROWS * 2) {                          // fc6 partial (fo half)
      int r = tid >> 1, c = tid & 1;
      float s = 0.0f;
      for (int k = 0; k < GRUH; ++k)
        s += (float)hn[(size_t)r * GRUH + k] * A.fc6w[c * 512 + k];
      PART[t * (ROWS * 2) + tid] = s;
    }
    dense(hn, GRUH, GRUH, A.fc3w, A.fc3b, 64, S1, 64, false, wave, lane); // f=fc3(hh)
    __syncthreads();
    f16* tm = hc; hc = hn; hn = tm;
  }
  // hc == h_last. Precompute constant gh for the backward cell:
  dense(hc, GRUH, GRUH, A.bwhh, A.bbhh, 768, GH, 768, false, wave, lane);
  for (int i = tid; i < ROWS * 64; i += 256) {     // b0 = fc5(g)
    int r = i >> 6, o = i & 63;
    S2[(size_t)r * 64 + o] =
        (f16)(A.fc5b[o] + G2[r * 2] * A.fc5w[o * 2] + G2[r * 2 + 1] * A.fc5w[o * 2 + 1]);
  }
  __syncthreads();

  // ---- backward rollout (12 steps), hidden input = h_last (reference bug)
  for (int j = 0; j < 12; ++j) {
    gru_step_pregh(S2, GH, hc, A.bwih, A.bbih, hn, wave, lane);  // bh -> hn
    __syncthreads();
    if (tid < ROWS * 2) {                          // out = fc6([fo, bh])
      int r = tid >> 1, c = tid & 1;
      float s = PART[(11 - j) * (ROWS * 2) + tid] + A.fc6b[c];
      for (int k = 0; k < GRUH; ++k)
        s += (float)hn[(size_t)r * GRUH + k] * A.fc6w[c * 512 + 256 + k];
      A.out_b[((size_t)j * NTOT + rowBase + r) * 2 + c] = s;
      G2[tid] = s;
    }
    __syncthreads();
    for (int i = tid; i < ROWS * 64; i += 256) {   // b = fc5(out)
      int r = i >> 6, o = i & 63;
      S2[(size_t)r * 64 + o] =
          (f16)(A.fc5b[o] + G2[r * 2] * A.fc5w[o * 2] + G2[r * 2 + 1] * A.fc5w[o * 2 + 1]);
    }
    __syncthreads();
  }
}

__global__ void cvt_f32_f16(const float* __restrict__ s, f16* __restrict__ d, int n) {
  int i = blockIdx.x * 256 + threadIdx.x;
  if (i < n) d[i] = (f16)s[i];
}

// d_in flattening (setup_inputs insertion order, params depth-first):
// 0 x, 1 y, 2 e, 3 h0x, 4 h0y, 5/6 fcx.w/b, 7/8 fcy.w/b,
// 9-12 grux(wih,whh,bih,bhh), 13-16 gruy, 17-22 prior(w,b)x3,
// 23-28 recognition, 29-34 goal, 35/36 fcf, 37/38 fc2, 39/40 fc3,
// 41/42 fc4(unused), 43/44 fc5, 45-48 forward_gru, 49-52 backward_gru, 53/54 fc6.
extern "C" void kernel_launch(void* const* d_in, const int* in_sizes, int n_in,
                              void* d_out, int out_size, void* d_ws, size_t ws_size,
                              hipStream_t stream) {
  (void)in_sizes; (void)n_in; (void)out_size; (void)ws_size;
  f16* ws = (f16*)d_ws;

  struct Cv { int idx; size_t off; int cnt; };
  static const Cv cv[] = {
      {9,       0,  49152}, {10,  49152, 196608}, {13, 245760,  49152},
      {14, 294912, 196608}, {17, 491520,  32768}, {19, 524288,   8192},
      {21, 532480,   4096}, {23, 536576, 131072}, {25, 667648,  32768},
      {27, 700416,   8192}, {29, 708608,  36864}, {31, 745472,   8192},
      {35, 753664,  18432}, {37, 772096,  73728}, {39, 845824,  16384},
      {45, 862208,  49152}, {46, 911360, 196608}, {49, 1107968, 49152},
      {50, 1157120, 196608}};
  for (const Cv& c : cv)
    cvt_f32_f16<<<(c.cnt + 255) / 256, 256, 0, stream>>>(
        (const float*)d_in[c.idx], ws + c.off, c.cnt);

  Args A;
  A.x = (const float*)d_in[0];  A.y = (const float*)d_in[1];
  A.e = (const float*)d_in[2];  A.h0x = (const float*)d_in[3];
  A.h0y = (const float*)d_in[4];
  A.fcxw = (const float*)d_in[5];  A.fcxb = (const float*)d_in[6];
  A.fcyw = (const float*)d_in[7];  A.fcyb = (const float*)d_in[8];
  A.gxwih = ws + 0;       A.gxwhh = ws + 49152;
  A.gxbih = (const float*)d_in[11]; A.gxbhh = (const float*)d_in[12];
  A.gywih = ws + 245760;  A.gywhh = ws + 294912;
  A.gybih = (const float*)d_in[15]; A.gybhh = (const float*)d_in[16];
  A.pr0w = ws + 491520; A.pr0b = (const float*)d_in[18];
  A.pr1w = ws + 524288; A.pr1b = (const float*)d_in[20];
  A.pr2w = ws + 532480; A.pr2b = (const float*)d_in[22];
  A.rc0w = ws + 536576; A.rc0b = (const float*)d_in[24];
  A.rc1w = ws + 667648; A.rc1b = (const float*)d_in[26];
  A.rc2w = ws + 700416; A.rc2b = (const float*)d_in[28];
  A.gl0w = ws + 708608; A.gl0b = (const float*)d_in[30];
  A.gl1w = ws + 745472; A.gl1b = (const float*)d_in[32];
  A.gl2w = (const float*)d_in[33]; A.gl2b = (const float*)d_in[34];
  A.fcfw = ws + 753664; A.fcfb = (const float*)d_in[36];
  A.fc2w = ws + 772096; A.fc2b = (const float*)d_in[38];
  A.fc3w = ws + 845824; A.fc3b = (const float*)d_in[40];
  A.fc5w = (const float*)d_in[43]; A.fc5b = (const float*)d_in[44];
  A.fwih = ws + 862208;  A.fwhh = ws + 911360;
  A.fbih = (const float*)d_in[47]; A.fbhh = (const float*)d_in[48];
  A.bwih = ws + 1107968; A.bwhh = ws + 1157120;
  A.bbih = (const float*)d_in[51]; A.bbhh = (const float*)d_in[52];
  A.fc6w = (const float*)d_in[53]; A.fc6b = (const float*)d_in[54];

  float* out = (float*)d_out;
  A.out_p = out;
  A.out_q = out + (size_t)NTOT * 64;
  A.out_g = out + (size_t)2 * NTOT * 64;
  A.out_b = out + (size_t)2 * NTOT * 64 + (size_t)NTOT * 2;

  const size_t shmem = SM_BYTES;  // ~169 KB of the 320 KB WGP LDS
  hipFuncSetAttribute((const void*)bitrap_fused,
                      hipFuncAttributeMaxDynamicSharedMemorySize, (int)shmem);
  bitrap_fused<<<NTOT / ROWS, 256, shmem, stream>>>(A);
}
